// DynamicsModel_54339926229264
// MI455X (gfx1250) — matrix operand order repaired and checked by reading, not compiled, not run on hardware
//
#include <hip/hip_runtime.h>
#include <hip/hip_bf16.h>

// RSSM scan: B=256, T=64, H=2048, S=512, A=64, E=2048
// 63 sequential steps; each step = 7 bf16-WMMA GEMMs (f32 accumulate) + GRU /
// rsample elementwise. fp32 weights (~168MB) stay L2-resident (192MB L2).
// GEMM is software-pipelined: tile kt+1 global loads (into VGPRs) are issued
// before tile kt's WMMA compute, so L2 latency hides behind matrix work.
// LDS tiles are pre-swizzled into WMMA fragment layout (ds_load_b128 reads);
// fp32->bf16 is 3 VALU ops per pair (2x v_add + v_perm_b32).

typedef __attribute__((ext_vector_type(16))) __bf16 v16bf;
typedef __attribute__((ext_vector_type(8)))  float  v8f;

#define Bdim 256
#define Tdim 64
#define Hdim 2048
#define Sdim 512
#define Adim 64
#define Edim 2048
#define Tm1  63

#define BM 64
#define BN 128
#define BK 32
#define LPAD 24   // ushorts per lane slot (16 used + 8 pad; 48B, 16B aligned)

union FragU {
    uint4 q[2];
    v16bf v;
};

// pack bf16(x) into [15:0], bf16(y) into [31:16]; round-half-up via +0x8000
__device__ __forceinline__ unsigned pack_bf16(float x, float y) {
    unsigned ux = __float_as_uint(x) + 0x8000u;
    unsigned uy = __float_as_uint(y) + 0x8000u;
    return __builtin_amdgcn_perm(uy, ux, 0x07060302u);
}

__device__ __forceinline__ float sigmoidf(float x) {
    return 1.0f / (1.0f + __expf(-x));
}

// ---- A tile (BM x BK): global -> regs ----
__device__ __forceinline__ void load_A(float4 va[2],
                                       const float* __restrict__ A1, int lda1,
                                       const float* __restrict__ A2, int lda2, int K1,
                                       int blockM, int k0, int tid)
{
    #pragma unroll
    for (int j = 0; j < 2; ++j) {
        int e = (tid + j * 256) * 4;
        int r = e >> 5;              // tile row
        int c = e & 31;              // tile k, multiple of 4
        int gk = k0 + c;
        const float* src = (gk < K1)
            ? &A1[(size_t)(blockM + r) * lda1 + gk]
            : &A2[(size_t)(blockM + r) * lda2 + (gk - K1)];
        va[j] = *(const float4*)src;
    }
}

// ---- A tile: regs -> fragment-layout LDS ----
// slot = rowTile*32 + (row&15) + 16*hi ; element i: lanes 0-15 hold
// K {0..15}, lanes 16-31 hold K {16..31}.
__device__ __forceinline__ void store_A(unsigned short* __restrict__ Abuf,
                                        const float4 va[2], int tid)
{
    #pragma unroll
    for (int j = 0; j < 2; ++j) {
        int e = (tid + j * 256) * 4;
        int r = e >> 5, c = e & 31;
        int hi = c >> 4;
        int ii = c & 15;                         // multiple of 4
        unsigned p0 = pack_bf16(va[j].x, va[j].y);
        unsigned p1 = pack_bf16(va[j].z, va[j].w);
        unsigned short* dst =
            &Abuf[(size_t)(((r >> 4) * 32) + (r & 15) + hi * 16) * LPAD + ii];
        *(uint2*)dst = make_uint2(p0, p1);
    }
}

// ---- B tile (BK x BN): global -> regs (with one-tile-ahead prefetch) ----
__device__ __forceinline__ void load_B(float4 vb[4],
                                       const float* __restrict__ Bmat, int Bld,
                                       int transB, int blockN, int k0, int K, int tid)
{
    if (!transB) {
        #pragma unroll
        for (int j = 0; j < 4; ++j) {
            int e = (tid + j * 256) * 4;
            int r = e >> 7;              // k
            int c = e & 127;             // n, multiple of 4
            const float* p = &Bmat[(size_t)(k0 + r) * Bld + (blockN + c)];
            vb[j] = *(const float4*)p;
            if (k0 + BK < K) __builtin_prefetch(p + (size_t)BK * Bld, 0, 1);
        }
    } else {
        #pragma unroll
        for (int j = 0; j < 4; ++j) {
            int e = (tid + j * 256) * 4;
            int c = e >> 5;              // n
            int r = e & 31;              // k, multiple of 4
            const float* p = &Bmat[(size_t)(blockN + c) * Bld + (k0 + r)];
            vb[j] = *(const float4*)p;
            if (k0 + BK < K) __builtin_prefetch(p + BK, 0, 1);
        }
    }
}

// ---- B tile: regs -> fragment-layout LDS ----
// slot = colTile*32 + (col&15) + 16*hi ; element = k&15.
__device__ __forceinline__ void store_B(unsigned short* __restrict__ Bbuf,
                                        const float4 vb[4], int transB, int tid)
{
    if (!transB) {
        #pragma unroll
        for (int j = 0; j < 4; ++j) {
            int e = (tid + j * 256) * 4;
            int r = e >> 7, c = e & 127;
            int hi = r >> 4, ii = r & 15;
            unsigned p0 = pack_bf16(vb[j].x, vb[j].y);
            unsigned p1 = pack_bf16(vb[j].z, vb[j].w);
            // 4 consecutive cols -> 4 different lane slots
            size_t base = (size_t)((c >> 4) * 32 + hi * 16) * LPAD + ii;
            Bbuf[base + (size_t)((c + 0) & 15) * LPAD] = (unsigned short)(p0);
            Bbuf[base + (size_t)((c + 1) & 15) * LPAD] = (unsigned short)(p0 >> 16);
            Bbuf[base + (size_t)((c + 2) & 15) * LPAD] = (unsigned short)(p1);
            Bbuf[base + (size_t)((c + 3) & 15) * LPAD] = (unsigned short)(p1 >> 16);
        }
    } else {
        #pragma unroll
        for (int j = 0; j < 4; ++j) {
            int e = (tid + j * 256) * 4;
            int c = e >> 5, r = e & 31;
            int hi = r >> 4;
            int ii = r & 15;             // multiple of 4
            unsigned p0 = pack_bf16(vb[j].x, vb[j].y);
            unsigned p1 = pack_bf16(vb[j].z, vb[j].w);
            unsigned short* dst =
                &Bbuf[(size_t)(((c >> 4) * 32) + (c & 15) + hi * 16) * LPAD + ii];
            *(uint2*)dst = make_uint2(p0, p1);
        }
    }
}

// C[M,N] = act( concat(A1,A2)[M,K] @ B + bias ), bf16 WMMA, f32 accumulate.
// B element (k,n) = transB ? Bmat[n*Bld + k] : Bmat[k*Bld + n].
// Requires M%64==0, N%128==0, K%32==0 (true for all call sites).
__global__ __launch_bounds__(256)
void gemm_bf16_wmma(const float* __restrict__ A1, int lda1,
                    const float* __restrict__ A2, int lda2, int K1,
                    const float* __restrict__ Bmat, int Bld, int transB,
                    const float* __restrict__ bias,
                    float* __restrict__ C, int ldc,
                    int N, int K, int relu)
{
    __shared__ __align__(16) unsigned short As[2][4 * 32 * LPAD];  // 12.0 KB
    __shared__ __align__(16) unsigned short Bs[2][8 * 32 * LPAD];  // 24.0 KB

    const int tid  = threadIdx.x;
    const int wave = tid >> 5;
    const int lane = tid & 31;
    const int l    = lane & 15;
    const int hi   = lane >> 4;
    const int wm   = wave & 1;           // 2 waves along M
    const int wn   = wave >> 1;          // 4 waves along N
    const int blockM = blockIdx.y * BM;
    const int blockN = blockIdx.x * BN;

    v8f acc[2][2];
    #pragma unroll
    for (int i = 0; i < 2; ++i)
        #pragma unroll
        for (int j = 0; j < 2; ++j)
            acc[i][j] = (v8f){0.f, 0.f, 0.f, 0.f, 0.f, 0.f, 0.f, 0.f};

    const int nk = K / BK;
    float4 va[2], vb[4];
    load_A(va, A1, lda1, A2, lda2, K1, blockM, 0, tid);
    load_B(vb, Bmat, Bld, transB, blockN, 0, K, tid);

    int p = 0;
    for (int kt = 0; kt < nk; ++kt) {
        // drain regs (tile kt) into LDS buf p -- single batched loadcnt wait
        store_A(As[p], va, tid);
        store_B(Bs[p], vb, transB, tid);
        // issue tile kt+1 global loads; latency hides behind compute below
        if (kt + 1 < nk) {
            load_A(va, A1, lda1, A2, lda2, K1, blockM, (kt + 1) * BK, tid);
            load_B(vb, Bmat, Bld, transB, blockN, (kt + 1) * BK, K, tid);
        }
        __syncthreads();

        // ---- fragment loads: two ds_load_b128 per fragment ----
        FragU af[2], bf_[2];
        #pragma unroll
        for (int fm = 0; fm < 2; ++fm) {
            const unsigned short* src = &As[p][(size_t)((wm * 2 + fm) * 32 + lane) * LPAD];
            af[fm].q[0] = *(const uint4*)(src);
            af[fm].q[1] = *(const uint4*)(src + 8);
        }
        #pragma unroll
        for (int fn = 0; fn < 2; ++fn) {
            const unsigned short* src = &Bs[p][(size_t)((wn * 2 + fn) * 32 + lane) * LPAD];
            bf_[fn].q[0] = *(const uint4*)(src);
            bf_[fn].q[1] = *(const uint4*)(src + 8);
        }

        #pragma unroll
        for (int fm = 0; fm < 2; ++fm)
            #pragma unroll
            for (int fn = 0; fn < 2; ++fn)
                acc[fm][fn] = __builtin_amdgcn_wmma_f32_16x16x32_bf16(
                    false, af[fm].v, false, bf_[fn].v,
                    (short)0, acc[fm][fn], false, false);

        p ^= 1;
    }

    // ---- epilogue: bias + optional relu, fp32 store ----
    #pragma unroll
    for (int fm = 0; fm < 2; ++fm) {
        #pragma unroll
        for (int fn = 0; fn < 2; ++fn) {
            int col = blockN + wn * 32 + fn * 16 + l;
            float bval = bias ? bias[col] : 0.0f;
            #pragma unroll
            for (int r = 0; r < 8; ++r) {
                int row = blockM + wm * 32 + fm * 16 + hi * 8 + r;
                float v = acc[fm][fn][r] + bval;
                if (relu) v = fmaxf(v, 0.0f);
                C[(size_t)row * ldc + col] = v;
            }
        }
    }
}

// h_new = (1-z)*tanh(in + r*hn) + z*h_prev ; gate order (r,z,n)
__global__ __launch_bounds__(256)
void gru_update(const float* __restrict__ gi, const float* __restrict__ gh,
                const float* __restrict__ hprev, int ldp,
                float* __restrict__ hout, int ldo)
{
    int idx = blockIdx.x * blockDim.x + threadIdx.x;
    if (idx >= Bdim * Hdim) return;
    int b = idx / Hdim, j = idx % Hdim;
    size_t g = (size_t)b * 3 * Hdim + j;
    float ir = gi[g], iz = gi[g + Hdim], inn = gi[g + 2 * Hdim];
    float hr = gh[g], hz = gh[g + Hdim], hn  = gh[g + 2 * Hdim];
    float r = sigmoidf(ir + hr);
    float z = sigmoidf(iz + hz);
    float n = tanhf(inn + r * hn);
    float h = hprev[(size_t)b * ldp + j];
    hout[(size_t)b * ldo + j] = (1.0f - z) * n + z * h;
}

// lin = [m | lv]; s = m + exp(softplus(lv)) * noise = m + (1+exp(lv)) * noise
__global__ __launch_bounds__(256)
void rsample(const float* __restrict__ lin,
             const float* __restrict__ noise, int ldn,
             float* __restrict__ mout, int ldm,
             float* __restrict__ lvout, int ldlv,
             float* __restrict__ sout, int lds_)
{
    int idx = blockIdx.x * blockDim.x + threadIdx.x;
    if (idx >= Bdim * Sdim) return;
    int b = idx / Sdim, j = idx % Sdim;
    float m  = lin[(size_t)b * 2 * Sdim + j];
    float lv = lin[(size_t)b * 2 * Sdim + Sdim + j];
    float scale = 1.0f + __expf(lv);
    mout[(size_t)b * ldm + j]   = m;
    lvout[(size_t)b * ldlv + j] = lv;
    sout[(size_t)b * lds_ + j]  = m + scale * noise[(size_t)b * ldn + j];
}

__global__ __launch_bounds__(256)
void init_state(const float* __restrict__ ph, const float* __restrict__ ps,
                float* __restrict__ hiddens, float* __restrict__ prior_s,
                float* __restrict__ post_s)
{
    int idx = blockIdx.x * blockDim.x + threadIdx.x;
    if (idx < Bdim * Hdim) {
        int b = idx / Hdim, j = idx % Hdim;
        hiddens[(size_t)b * Tdim * Hdim + j] = ph[idx];
    }
    if (idx < Bdim * Sdim) {
        int b = idx / Sdim, j = idx % Sdim;
        float v = ps[idx];
        prior_s[(size_t)b * Tdim * Sdim + j] = v;
        post_s[(size_t)b * Tdim * Sdim + j]  = v;
    }
}

extern "C" void kernel_launch(void* const* d_in, const int* in_sizes, int n_in,
                              void* d_out, int out_size, void* d_ws, size_t ws_size,
                              hipStream_t stream) {
    const float* prev_hidden = (const float*)d_in[0];
    const float* prev_state  = (const float*)d_in[1];
    const float* actions     = (const float*)d_in[2];
    const float* obs         = (const float*)d_in[3];
    const float* prior_noise = (const float*)d_in[4];
    const float* post_noise  = (const float*)d_in[5];
    const float* W_sa = (const float*)d_in[6];  const float* b_sa = (const float*)d_in[7];
    const float* W_ih = (const float*)d_in[8];  const float* W_hh = (const float*)d_in[9];
    const float* b_ih = (const float*)d_in[10]; const float* b_hh = (const float*)d_in[11];
    const float* W_ha = (const float*)d_in[12]; const float* b_ha = (const float*)d_in[13];
    const float* W_pr = (const float*)d_in[14]; const float* b_pr = (const float*)d_in[15];
    const float* W_ho = (const float*)d_in[16]; const float* b_ho = (const float*)d_in[17];
    const float* W_po = (const float*)d_in[18]; const float* b_po = (const float*)d_in[19];

    float* out = (float*)d_out;
    float* hiddens      = out;                                   // [B,T,H]
    float* prior_states = hiddens + (size_t)Bdim * Tdim * Hdim;  // [B,T,S]
    float* post_states  = prior_states + (size_t)Bdim * Tdim * Sdim;
    float* pm_out  = post_states + (size_t)Bdim * Tdim * Sdim;   // [B,T-1,S]
    float* plv_out = pm_out  + (size_t)Bdim * Tm1 * Sdim;
    float* qm_out  = plv_out + (size_t)Bdim * Tm1 * Sdim;
    float* qlv_out = qm_out  + (size_t)Bdim * Tm1 * Sdim;

    float* ws = (float*)d_ws;                 // ~20MB of fp32 scratch
    float* sa = ws;                           // [B, H]
    float* gi = sa + (size_t)Bdim * Hdim;     // [B, 3H]
    float* gh = gi + (size_t)Bdim * 3 * Hdim; // [B, 3H]
    float* ha = gh + (size_t)Bdim * 3 * Hdim; // [B, H]
    float* ho = ha + (size_t)Bdim * Hdim;     // [B, H]
    float* tp = ho + (size_t)Bdim * Hdim;     // [B, 2S]
    float* tq = tp + (size_t)Bdim * 2 * Sdim; // [B, 2S]

    init_state<<<(Bdim * Hdim + 255) / 256, 256, 0, stream>>>(
        prev_hidden, prev_state, hiddens, prior_states, post_states);

    const dim3 blk(256);
    for (int t = 0; t < Tm1; ++t) {
        const float* s_t = post_states + (size_t)t * Sdim;   // stride T*S
        const float* a_t = actions + (size_t)t * Adim;       // stride T*A
        const float* o_t = obs + (size_t)t * Edim;           // stride T*E
        const float* h_t = hiddens + (size_t)t * Hdim;       // stride T*H
        float* h_t1 = hiddens + (size_t)(t + 1) * Hdim;

        // sa = relu([s,a] @ W_sa + b_sa)            [B, H]
        gemm_bf16_wmma<<<dim3(Hdim / BN, Bdim / BM), blk, 0, stream>>>(
            s_t, Tdim * Sdim, a_t, Tdim * Adim, Sdim,
            W_sa, Hdim, 0, b_sa, sa, Hdim, Hdim, Sdim + Adim, 1);
        // gi = sa @ W_ih^T + b_ih                   [B, 3H]
        gemm_bf16_wmma<<<dim3(3 * Hdim / BN, Bdim / BM), blk, 0, stream>>>(
            sa, Hdim, nullptr, 0, Hdim,
            W_ih, Hdim, 1, b_ih, gi, 3 * Hdim, 3 * Hdim, Hdim, 0);
        // gh = h @ W_hh^T + b_hh                    [B, 3H]
        gemm_bf16_wmma<<<dim3(3 * Hdim / BN, Bdim / BM), blk, 0, stream>>>(
            h_t, Tdim * Hdim, nullptr, 0, Hdim,
            W_hh, Hdim, 1, b_hh, gh, 3 * Hdim, 3 * Hdim, Hdim, 0);
        // h_{t+1} = GRU(gi, gh, h_t)
        gru_update<<<(Bdim * Hdim + 255) / 256, blk, 0, stream>>>(
            gi, gh, h_t, Tdim * Hdim, h_t1, Tdim * Hdim);
        // ha = relu([h,a] @ W_ha + b_ha)            [B, H]
        gemm_bf16_wmma<<<dim3(Hdim / BN, Bdim / BM), blk, 0, stream>>>(
            h_t1, Tdim * Hdim, a_t, Tdim * Adim, Hdim,
            W_ha, Hdim, 0, b_ha, ha, Hdim, Hdim, Hdim + Adim, 1);
        // prior logits = ha @ W_prior + b_prior     [B, 2S]
        gemm_bf16_wmma<<<dim3(2 * Sdim / BN, Bdim / BM), blk, 0, stream>>>(
            ha, Hdim, nullptr, 0, Hdim,
            W_pr, 2 * Sdim, 0, b_pr, tp, 2 * Sdim, 2 * Sdim, Hdim, 0);
        rsample<<<(Bdim * Sdim + 255) / 256, blk, 0, stream>>>(
            tp, prior_noise + (size_t)t * Sdim, Tm1 * Sdim,
            pm_out + (size_t)t * Sdim, Tm1 * Sdim,
            plv_out + (size_t)t * Sdim, Tm1 * Sdim,
            prior_states + (size_t)(t + 1) * Sdim, Tdim * Sdim);
        // ho = relu([h,o] @ W_ho + b_ho)            [B, H]
        gemm_bf16_wmma<<<dim3(Hdim / BN, Bdim / BM), blk, 0, stream>>>(
            h_t1, Tdim * Hdim, o_t, Tdim * Edim, Hdim,
            W_ho, Hdim, 0, b_ho, ho, Hdim, Hdim, Hdim + Edim, 1);
        // post logits = ho @ W_post + b_post        [B, 2S]
        gemm_bf16_wmma<<<dim3(2 * Sdim / BN, Bdim / BM), blk, 0, stream>>>(
            ho, Hdim, nullptr, 0, Hdim,
            W_po, 2 * Sdim, 0, b_po, tq, 2 * Sdim, 2 * Sdim, Hdim, 0);
        rsample<<<(Bdim * Sdim + 255) / 256, blk, 0, stream>>>(
            tq, post_noise + (size_t)t * Sdim, Tm1 * Sdim,
            qm_out + (size_t)t * Sdim, Tm1 * Sdim,
            qlv_out + (size_t)t * Sdim, Tm1 * Sdim,
            post_states + (size_t)(t + 1) * Sdim, Tdim * Sdim);
    }
    (void)in_sizes; (void)n_in; (void)out_size; (void)ws_size;
}
